// TransformerBlock_3118146257340
// MI455X (gfx1250) — compile-verified, matrix operand order checked by reading, not run on hardware
//
#include <hip/hip_runtime.h>
#include <hip/hip_bf16.h>
#include <math.h>

// ---------------------------------------------------------------------------
// Transformer block (B=1, S=4096, D=768, H=12, HD=64, FF=3072) on gfx1250.
// ~110 GFLOP vs ~55MB minimum traffic -> strongly compute bound: bf16 WMMA
// (v_wmma_f32_16x16x32_bf16, fp32 accum) everywhere, flash-attention so the
// 805MB score tensor never exists, and GLOBAL_LOAD_ASYNC_TO_LDS_B128 (ASYNCcnt)
// staging with double buffering so HBM->LDS copies overlap the matrix pipe.
// ---------------------------------------------------------------------------

typedef __attribute__((ext_vector_type(16))) __bf16 v16bf;
typedef __attribute__((ext_vector_type(8)))  __bf16 v8bf;
typedef __attribute__((ext_vector_type(8)))  float  v8f;

constexpr int S_  = 4096;
constexpr int D_  = 768;
constexpr int H_  = 12;
constexpr int HD_ = 64;
constexpr int FF_ = 3072;

__device__ __forceinline__ v16bf cat16(v8bf lo, v8bf hi) {
  return __builtin_shufflevector(lo, hi, 0,1,2,3,4,5,6,7,8,9,10,11,12,13,14,15);
}
// A/B fragment loader: p points at a 32-wide K-run; lane half hl selects
// K offsets {hl*8..+8} and {16+hl*8..+8} per the ISA 16-bit operand layout.
__device__ __forceinline__ v16bf load_frag(const __bf16* p, int hl) {
  v8bf lo = *reinterpret_cast<const v8bf*>(p + hl * 8);
  v8bf hi = *reinterpret_cast<const v8bf*>(p + 16 + hl * 8);
  return cat16(lo, hi);
}
__device__ __forceinline__ v8f wmma_bf16(v16bf a, v16bf b, v8f c) {
  return __builtin_amdgcn_wmma_f32_16x16x32_bf16(false, a, false, b, (short)0, c,
                                                 false, false);
}

// CDNA5 async global->LDS copy (no VGPR round trip, tracked by ASYNCcnt).
// Generic shared pointers carry the LDS offset in their low 32 bits
// (ISA 10.2); the LDS address comes from the VGPR operand plus the offset.
__device__ __forceinline__ void async_copy_b128(void* lds_dst, const void* gsrc) {
  const unsigned lds = (unsigned)(uintptr_t)lds_dst;
  asm volatile("global_load_async_to_lds_b128 %0, %1, off"
               :: "v"(lds), "v"(gsrc) : "memory");
}
__device__ __forceinline__ void async_wait0() {
  asm volatile("s_wait_asynccnt 0x0" ::: "memory");
}

// ---------------------------------------------------------------------------
// Weight prep: [K][N] fp32 -> [N][K] bf16 (transpose through an LDS tile).
// Done once per call; lets GEMM B-staging be straight contiguous async copies.
// ---------------------------------------------------------------------------
__global__ __launch_bounds__(256) void transpose_to_bf16_kernel(
    const float* __restrict__ in, __bf16* __restrict__ out, int K, int N) {
  __shared__ float tile[32][33];
  const int kb = blockIdx.y * 32, nb = blockIdx.x * 32;
  const int tx = threadIdx.x & 31, ty = threadIdx.x >> 5;   // 32 x 8
  #pragma unroll
  for (int yy = 0; yy < 4; ++yy) {
    int k = ty + yy * 8;
    tile[k][tx] = in[(size_t)(kb + k) * N + nb + tx];
  }
  __syncthreads();
  #pragma unroll
  for (int yy = 0; yy < 4; ++yy) {
    int n = ty + yy * 8;
    out[(size_t)(nb + n) * K + kb + tx] = (__bf16)tile[tx][n];
  }
}

// ---------------------------------------------------------------------------
// LayerNorm (fp32 in) -> bf16 out.  One block per row of 768.
// ---------------------------------------------------------------------------
__global__ __launch_bounds__(256) void ln_bf16_kernel(
    const float* __restrict__ x, const float* __restrict__ w,
    const float* __restrict__ b, __bf16* __restrict__ out) {
  __shared__ float red[256];
  const int row = blockIdx.x;
  const float* xr = x + (size_t)row * D_;
  float s = 0.f;
  for (int j = threadIdx.x; j < D_; j += 256) s += xr[j];
  red[threadIdx.x] = s; __syncthreads();
  for (int off = 128; off > 0; off >>= 1) {
    if (threadIdx.x < off) red[threadIdx.x] += red[threadIdx.x + off];
    __syncthreads();
  }
  const float mu = red[0] * (1.0f / D_);
  __syncthreads();
  float vs = 0.f;
  for (int j = threadIdx.x; j < D_; j += 256) { float d = xr[j] - mu; vs += d * d; }
  red[threadIdx.x] = vs; __syncthreads();
  for (int off = 128; off > 0; off >>= 1) {
    if (threadIdx.x < off) red[threadIdx.x] += red[threadIdx.x + off];
    __syncthreads();
  }
  const float rstd = rsqrtf(red[0] * (1.0f / D_) + 1e-5f);
  for (int j = threadIdx.x; j < D_; j += 256)
    out[(size_t)row * D_ + j] = (__bf16)((xr[j] - mu) * rstd * w[j] + b[j]);
}

// ---------------------------------------------------------------------------
// Tiled bf16 GEMM: C[M,N] = A[M,K] * Bt[N,K]^T (+bias epilogues)
// Block = 128 threads (4 waves), tile 64x64, K-step 32, double-buffered LDS
// fed by async copies. Each wave owns a 32x32 quadrant: 4 WMMA / K-step with
// 2 A-frags x 2 B-frags reuse.
// EPI 0: outB = bf16((acc+bias)*scale)      (Q/K projections; Q scale 0.125)
// EPI 1: outF = acc + bias + resid          (O-proj, down-proj)
// EPI 2: outB = bf16(gelu_exact(acc+bias))  (up-proj)
// EPI 3: outB[col*S_+row] = acc + bias      (V projection, stored transposed)
// ---------------------------------------------------------------------------
template <int EPI>
__global__ __launch_bounds__(128) void gemm_bf16_kernel(
    const __bf16* __restrict__ A, const __bf16* __restrict__ Bt,
    const float* __restrict__ bias, const float* __restrict__ resid,
    float* __restrict__ outF, __bf16* __restrict__ outB,
    int M, int N, int K, float scale) {
  __shared__ alignas(16) __bf16 As[2][64][32];   // [m][k]
  __shared__ alignas(16) __bf16 Bs[2][64][32];   // [n][k]

  const int n0 = blockIdx.x * 64;
  const int m0 = blockIdx.y * 64;
  const int tid  = threadIdx.x;
  const int wave = tid >> 5, lane = tid & 31;
  const int wr = (wave & 1) * 32;     // wave row base (2 sub-tiles of 16)
  const int wc = (wave >> 1) * 32;    // wave col base (2 sub-tiles of 16)
  const int c  = lane & 15, hl = lane >> 4;
  const int sr = tid >> 1, sg = (tid & 1) << 4;   // staging: 64 rows x 2 halves

  auto stage = [&](int buf, int kk) {
    const __bf16* ga = A + (size_t)(m0 + sr) * K + kk + sg;
    async_copy_b128(&As[buf][sr][sg],     ga);
    async_copy_b128(&As[buf][sr][sg + 8], ga + 8);
    const __bf16* gb = Bt + (size_t)(n0 + sr) * K + kk + sg;
    async_copy_b128(&Bs[buf][sr][sg],     gb);
    async_copy_b128(&Bs[buf][sr][sg + 8], gb + 8);
  };

  v8f acc[2][2] = {};
  stage(0, 0);
  async_wait0();
  __syncthreads();

  int buf = 0;
  for (int kk = 0; kk < K; kk += 32) {
    if (kk + 32 < K) stage(buf ^ 1, kk + 32);   // prefetch overlaps the WMMAs
    v16bf a[2], b[2];
    #pragma unroll
    for (int i = 0; i < 2; ++i) {
      a[i] = load_frag(&As[buf][wr + i * 16 + c][0], hl);
      b[i] = load_frag(&Bs[buf][wc + i * 16 + c][0], hl);
    }
    #pragma unroll
    for (int i = 0; i < 2; ++i)
      #pragma unroll
      for (int j = 0; j < 2; ++j)
        acc[i][j] = wmma_bf16(a[i], b[j], acc[i][j]);
    async_wait0();
    __syncthreads();
    buf ^= 1;
  }

  #pragma unroll
  for (int i = 0; i < 2; ++i)
    #pragma unroll
    for (int j = 0; j < 2; ++j)
      #pragma unroll
      for (int e = 0; e < 8; ++e) {
        const int row = m0 + wr + i * 16 + hl * 8 + e;   // C layout: VGPR e + half
        const int col = n0 + wc + j * 16 + c;            // lane % 16
        float v = acc[i][j][e] + bias[col];
        if constexpr (EPI == 0) {
          outB[(size_t)row * N + col] = (__bf16)(v * scale);
        } else if constexpr (EPI == 1) {
          outF[(size_t)row * N + col] = v + resid[(size_t)row * N + col];
        } else if constexpr (EPI == 2) {
          outB[(size_t)row * N + col] =
              (__bf16)(0.5f * v * (1.0f + erff(v * 0.70710678118f)));
        } else {
          outB[(size_t)col * S_ + row] = (__bf16)v;      // vT[dim][seq]
        }
      }
}

// ---------------------------------------------------------------------------
// Flash attention: grid (S/64, H), block 128 (4 waves).
// Wave owns a 16-row query tile; 32 keys streamed per iteration. K tile and
// pre-transposed V tile arrive via async copies. Q pre-scaled by 1/sqrt(HD).
// ---------------------------------------------------------------------------
__global__ __launch_bounds__(128) void attn_kernel(
    const __bf16* __restrict__ Q, const __bf16* __restrict__ Kb,
    const __bf16* __restrict__ vT,               // [D][S] transposed values
    const int* __restrict__ amask, __bf16* __restrict__ ctx) {
  __shared__ alignas(16) __bf16 Ks[32][64];      // [key][dim]  (QK^T B-frags)
  __shared__ alignas(16) __bf16 Vt[64][32];      // [dim][key]  (PV  B-frags)
  __shared__ alignas(16) __bf16 Pl[4][16][32];   // per-wave P transpose buffer

  const int tid  = threadIdx.x;
  const int wave = tid >> 5, lane = tid & 31;
  const int c  = lane & 15, hl = lane >> 4;
  const int h    = blockIdx.y;
  const int q0   = blockIdx.x * 64 + wave * 16;
  const int hoff = h * HD_;

  v16bf aq[2];
  #pragma unroll
  for (int kc = 0; kc < 2; ++kc)
    aq[kc] = load_frag(Q + (size_t)(q0 + c) * D_ + hoff + kc * 32, hl);

  v8f o[4] = {};
  float m[8], l[8];
  #pragma unroll
  for (int i = 0; i < 8; ++i) { m[i] = -1e30f; l[i] = 0.f; }

  const int kend = (blockIdx.x + 1) * 64;        // causal bound (block-uniform)
  for (int kb = 0; kb < kend; kb += 32) {
    __syncthreads();                             // protect LDS reuse
    {   // async-stage 32 keys of K (row-major) and V (already transposed)
      const int key = tid >> 2, db = (tid & 3) << 4;
      const __bf16* gk = Kb + (size_t)(kb + key) * D_ + hoff + db;
      async_copy_b128(&Ks[key][db],     gk);
      async_copy_b128(&Ks[key][db + 8], gk + 8);
      const int dv = tid >> 1, kg = (tid & 1) << 4;
      const __bf16* gv = vT + (size_t)(hoff + dv) * S_ + kb + kg;
      async_copy_b128(&Vt[dv][kg],     gv);
      async_copy_b128(&Vt[dv][kg + 8], gv + 8);
    }
    async_wait0();
    __syncthreads();

    // S(16x32) = Q(16x64) @ K^T(64x32)
    v8f s[2] = {};
    #pragma unroll
    for (int sub = 0; sub < 2; ++sub)
      #pragma unroll
      for (int kc = 0; kc < 2; ++kc) {
        v16bf b = load_frag(&Ks[sub * 16 + c][kc * 32], hl);
        s[sub] = wmma_bf16(aq[kc], b, s[sub]);
      }

    const int mk0 = amask[kb + c];
    const int mk1 = amask[kb + 16 + c];

    // online softmax; row = hl*8 + i spans 16 lanes -> xor-shuffle reductions
    #pragma unroll
    for (int i = 0; i < 8; ++i) {
      const int qr = q0 + hl * 8 + i;
      if (kb + c > qr      || mk0 == 0) s[0][i] = -1e30f;
      if (kb + 16 + c > qr || mk1 == 0) s[1][i] = -1e30f;
      float mx = fmaxf(s[0][i], s[1][i]);
      mx = fmaxf(mx, __shfl_xor(mx, 1));
      mx = fmaxf(mx, __shfl_xor(mx, 2));
      mx = fmaxf(mx, __shfl_xor(mx, 4));
      mx = fmaxf(mx, __shfl_xor(mx, 8));
      const float mn  = fmaxf(m[i], mx);
      const float fac = __expf(m[i] - mn);
      m[i] = mn;
      const float p0 = __expf(s[0][i] - mn);
      const float p1 = __expf(s[1][i] - mn);
      float r = p0 + p1;
      r += __shfl_xor(r, 1); r += __shfl_xor(r, 2);
      r += __shfl_xor(r, 4); r += __shfl_xor(r, 8);
      l[i] = l[i] * fac + r;
      #pragma unroll
      for (int d = 0; d < 4; ++d) o[d][i] *= fac;
      Pl[wave][hl * 8 + i][c]      = (__bf16)p0;   // C-layout -> LDS
      Pl[wave][hl * 8 + i][16 + c] = (__bf16)p1;
    }

    // P as an A-fragment (lane = row), then O += P @ V
    v16bf ap = load_frag(&Pl[wave][c][0], hl);
    #pragma unroll
    for (int d = 0; d < 4; ++d) {
      v16bf bv = load_frag(&Vt[d * 16 + c][0], hl);
      o[d] = wmma_bf16(ap, bv, o[d]);
    }
  }

  #pragma unroll
  for (int i = 0; i < 8; ++i) {
    const float inv = 1.0f / l[i];
    const int row = q0 + hl * 8 + i;
    #pragma unroll
    for (int d = 0; d < 4; ++d)
      ctx[(size_t)row * D_ + hoff + d * 16 + c] = (__bf16)(o[d][i] * inv);
  }
}

// ---------------------------------------------------------------------------
// Host launcher
// ---------------------------------------------------------------------------
extern "C" void kernel_launch(void* const* d_in, const int* in_sizes, int n_in,
                              void* d_out, int out_size, void* d_ws, size_t ws_size,
                              hipStream_t stream) {
  const float* hidden = (const float*)d_in[0];
  const int*   mask   = (const int*)  d_in[1];
  const float* ln1w   = (const float*)d_in[2];
  const float* ln1b   = (const float*)d_in[3];
  const float* wq     = (const float*)d_in[4];
  const float* bq     = (const float*)d_in[5];
  const float* wk     = (const float*)d_in[6];
  const float* bk     = (const float*)d_in[7];
  const float* wv     = (const float*)d_in[8];
  const float* bv     = (const float*)d_in[9];
  const float* wo     = (const float*)d_in[10];
  const float* bo     = (const float*)d_in[11];
  const float* ln2w   = (const float*)d_in[12];
  const float* ln2b   = (const float*)d_in[13];
  const float* wup    = (const float*)d_in[14];
  const float* bup    = (const float*)d_in[15];
  const float* wdown  = (const float*)d_in[16];
  const float* bdown  = (const float*)d_in[17];
  float* out = (float*)d_out;

  constexpr size_t W_DD  = (size_t)D_ * D_;
  constexpr size_t W_DF  = (size_t)D_ * FF_;
  constexpr size_t ACT   = (size_t)S_ * D_;
  constexpr size_t ACTFF = (size_t)S_ * FF_;

  char* ws = (char*)d_ws;
  size_t off = 0;
  __bf16* wqT  = (__bf16*)(ws + off); off += W_DD * 2;   // all weights [N][K] bf16
  __bf16* wkT  = (__bf16*)(ws + off); off += W_DD * 2;
  __bf16* wvT  = (__bf16*)(ws + off); off += W_DD * 2;
  __bf16* woT  = (__bf16*)(ws + off); off += W_DD * 2;
  __bf16* wupT = (__bf16*)(ws + off); off += W_DF * 2;
  __bf16* wdnT = (__bf16*)(ws + off); off += W_DF * 2;
  __bf16* xln_bf  = (__bf16*)(ws + off); off += ACT * 2;
  __bf16* q_bf    = (__bf16*)(ws + off); off += ACT * 2;
  __bf16* k_bf    = (__bf16*)(ws + off); off += ACT * 2;
  __bf16* vT_bf   = (__bf16*)(ws + off); off += ACT * 2;  // [D][S]
  __bf16* ctx_bf  = (__bf16*)(ws + off); off += ACT * 2;
  float*  hidden2 = (float*) (ws + off); off += ACT * 4;
  __bf16* yln_bf  = (__bf16*)(ws + off); off += ACT * 2;
  __bf16* ff_bf   = (__bf16*)(ws + off); off += ACTFF * 2;

  auto transp = [&](const float* src, __bf16* dst, int K, int N) {
    transpose_to_bf16_kernel<<<dim3(N / 32, K / 32), 256, 0, stream>>>(src, dst, K, N);
  };
  transp(wq, wqT, D_, D_);   transp(wk, wkT, D_, D_);  transp(wv, wvT, D_, D_);
  transp(wo, woT, D_, D_);   transp(wup, wupT, D_, FF_);
  transp(wdown, wdnT, FF_, D_);

  // LN1
  ln_bf16_kernel<<<S_, 256, 0, stream>>>(hidden, ln1w, ln1b, xln_bf);

  // Projections (Q pre-scaled by 1/sqrt(HD); V written transposed)
  dim3 gDD(D_ / 64, S_ / 64);
  gemm_bf16_kernel<0><<<gDD, 128, 0, stream>>>(xln_bf, wqT, bq, nullptr, nullptr, q_bf,
                                               S_, D_, D_, 0.125f);
  gemm_bf16_kernel<0><<<gDD, 128, 0, stream>>>(xln_bf, wkT, bk, nullptr, nullptr, k_bf,
                                               S_, D_, D_, 1.0f);
  gemm_bf16_kernel<3><<<gDD, 128, 0, stream>>>(xln_bf, wvT, bv, nullptr, nullptr, vT_bf,
                                               S_, D_, D_, 1.0f);

  // Flash attention
  attn_kernel<<<dim3(S_ / 64, H_), 128, 0, stream>>>(q_bf, k_bf, vT_bf, mask, ctx_bf);

  // O-projection + residual -> hidden2 (fp32)
  gemm_bf16_kernel<1><<<gDD, 128, 0, stream>>>(ctx_bf, woT, bo, hidden, hidden2, nullptr,
                                               S_, D_, D_, 1.0f);

  // LN2
  ln_bf16_kernel<<<S_, 256, 0, stream>>>(hidden2, ln2w, ln2b, yln_bf);

  // Up-projection + exact GELU -> bf16
  gemm_bf16_kernel<2><<<dim3(FF_ / 64, S_ / 64), 128, 0, stream>>>(
      yln_bf, wupT, bup, nullptr, nullptr, ff_bf, S_, FF_, D_, 1.0f);

  // Down-projection + residual -> d_out (fp32)
  gemm_bf16_kernel<1><<<gDD, 128, 0, stream>>>(ff_bf, wdnT, bdown, hidden2, out, nullptr,
                                               S_, D_, FF_, 1.0f);
}
